// SimplifiedHGTConv_86139864089177
// MI455X (gfx1250) — matrix operand order repaired and checked by reading, not compiled, not run on hardware
//
#include <hip/hip_runtime.h>
#include <stddef.h>

// ---------------------------------------------------------------------------
// SimplifiedHGTConv for MI455X (gfx1250).
// All GEMMs via v_wmma_f32_16x16x32_bf16 (f32 in/out, bf16 operands, f32 acc).
// M_TILE=64 rows per WG to amortize weight reads (4x less L2 traffic than 16).
// x-tile staging uses global_load_async_to_lds_b128 (ASYNCcnt) when available.
// Deterministic mean reduction (two-stage partial sums, no float atomics).
// ---------------------------------------------------------------------------

#define NROW 65536
#define NT   1024          // NROW / 64 row-tiles

#if defined(__HIP_DEVICE_COMPILE__) && defined(__has_builtin)
#if __has_builtin(__builtin_amdgcn_global_load_async_to_lds_b128) && \
    __has_builtin(__builtin_amdgcn_s_wait_asynccnt)
#define HAVE_ASYNC_LDS 1
#endif
#endif
#ifndef HAVE_ASYNC_LDS
#define HAVE_ASYNC_LDS 0
#endif

typedef __attribute__((ext_vector_type(16))) __bf16       v16bf;
typedef __attribute__((ext_vector_type(8)))  float        v8f;
typedef __attribute__((ext_vector_type(4)))  float        f32x4;
typedef __attribute__((ext_vector_type(4)))  unsigned int u32x4;
typedef __attribute__((ext_vector_type(4)))  int          v4i;

#if HAVE_ASYNC_LDS
typedef __attribute__((address_space(1))) v4i* gl_v4i_p;   // global int4*
typedef __attribute__((address_space(3))) v4i* ld_v4i_p;   // LDS int4*
#endif

union ABFrag { v16bf v; u32x4 u[2]; };

__device__ __forceinline__ v8f wmma_bf16(v16bf a, v16bf b, v8f c) {
  return __builtin_amdgcn_wmma_f32_16x16x32_bf16(
      /*neg_a=*/false, a, /*neg_b=*/false, b,
      /*c_mod=*/(short)0, c, /*reuse_a=*/false, /*reuse_b=*/false);
}

// A fragment (16x32 bf16) from row-major LDS tile (stride 256 bf16):
// lanes 0-15: K = k0+0..7 and k0+16..23 ; lanes 16-31: +8   (ISA 7.12.2)
__device__ __forceinline__ v16bf afrag_lds(const __bf16* base, int l16, int half, int k0) {
  ABFrag r;
  const __bf16* p = base + l16 * 256 + k0 + half * 8;
  r.u[0] = *(const u32x4*)p;
  r.u[1] = *(const u32x4*)(p + 16);
  return r.v;
}

// B fragment (32x16 bf16) from row-major weight W[dout=256][din=256]:
// lane n = column n; lanes 0-15: K = k0+0..7 and k0+16..23 ; lanes 16-31: +8
__device__ __forceinline__ v16bf bfrag_w(const __bf16* W, int ncol, int half, int k0) {
  ABFrag r;
  const __bf16* p = W + ncol * 256 + k0 + half * 8;
  r.u[0] = *(const u32x4*)p;
  r.u[1] = *(const u32x4*)(p + 16);
  return r.v;
}

// C[64x256] = A[64x256] @ W^T ; wave computes col-tiles (ct0, ct0+1) for 4 row
// tiles; B fragments are loaded once per k-step and reused across row tiles.
__device__ __forceinline__ void gemm_tile64(const __bf16* A, const __bf16* W,
                                            int l16, int half, int ct0,
                                            v8f acc[4][2]) {
#pragma unroll
  for (int ks = 0; ks < 8; ++ks) {
    const int k0 = ks * 32;
    v16bf b0 = bfrag_w(W, ct0 * 16 + l16, half, k0);
    v16bf b1 = bfrag_w(W, ct0 * 16 + 16 + l16, half, k0);
#pragma unroll
    for (int rt = 0; rt < 4; ++rt) {
      v16bf a = afrag_lds(A + rt * 16 * 256, l16, half, k0);
      acc[rt][0] = wmma_bf16(a, b0, acc[rt][0]);
      acc[rt][1] = wmma_bf16(a, b1, acc[rt][1]);
    }
  }
}

// Stage 64 rows x 256 cols of f32 x into a bf16 LDS tile.
// Preferred path: async copy to LDS (ASYNCcnt) then LDS->LDS convert.
__device__ __forceinline__ void stage_x64(const float* x, __bf16* xb, float* xf, int tid) {
#if HAVE_ASYNC_LDS
#pragma unroll
  for (int i = 0; i < 16; ++i) {
    const size_t e = (size_t)(i * 256 + tid) * 4;   // float index of a 16B chunk
    __builtin_amdgcn_global_load_async_to_lds_b128(
        (gl_v4i_p)(x + e), (ld_v4i_p)(xf + e), 0, 0);
  }
  __builtin_amdgcn_s_wait_asynccnt(0);
  __syncthreads();
#pragma unroll 8
  for (int i = 0; i < 64; ++i) {
    const int idx = i * 256 + tid;
    xb[idx] = (__bf16)xf[idx];
  }
#else
  (void)xf;
#pragma unroll
  for (int i = 0; i < 16; ++i) {
    const int idx4 = i * 256 + tid;
    f32x4 v = ((const f32x4*)x)[idx4];
    const int b = idx4 * 4;
    xb[b + 0] = (__bf16)v.x;
    xb[b + 1] = (__bf16)v.y;
    xb[b + 2] = (__bf16)v.z;
    xb[b + 3] = (__bf16)v.w;
  }
#endif
}

// ---------------------------------------------------------------------------
struct P1 { const float* src[9]; __bf16* dst; };

__global__ __launch_bounds__(256) void k1_cvtw(P1 p) {
  const int idx = blockIdx.x * 256 + threadIdx.x;   // < 9*65536
  const int mat = idx >> 16, off = idx & 65535;
  p.dst[idx] = (__bf16)p.src[mat][off];
}

// ---------------------------------------------------------------------------
struct P2 { const float* x[5]; const float* bias[5]; const __bf16* wt; float* partial; };

__global__ __launch_bounds__(256) void k2_projsum(P2 p) {
  __shared__ __attribute__((aligned(16))) unsigned char smem[96 * 1024];
  __bf16* xb = (__bf16*)smem;                 // 32 KB [64][256] bf16
  float*  xf = (float*)(smem + 32 * 1024);    // 64 KB async stage

  const int tid = threadIdx.x, m = blockIdx.x, t = blockIdx.y;
  const int lane = tid & 31, wave = tid >> 5;
  const int half = lane >> 4, l16 = lane & 15, ct0 = wave * 2;

  stage_x64(p.x[t] + (size_t)m * 64 * 256, xb, xf, tid);
  __syncthreads();

  v8f acc[4][2] = {};
  gemm_tile64(xb, p.wt + (size_t)t * 65536, l16, half, ct0, acc);

  // per-column sums of this 64-row tile (bias folded analytically: 64*b[col])
  float s0 = 0.f, s1 = 0.f;
#pragma unroll
  for (int rt = 0; rt < 4; ++rt)
#pragma unroll
    for (int j = 0; j < 8; ++j) { s0 += acc[rt][0][j]; s1 += acc[rt][1][j]; }
  s0 += __shfl_xor(s0, 16);
  s1 += __shfl_xor(s1, 16);
  if (half == 0) {
    const int c0 = ct0 * 16 + l16, c1 = c0 + 16;
    p.partial[((size_t)t * 256 + c0) * NT + m] = s0 + 64.f * p.bias[t][c0];
    p.partial[((size_t)t * 256 + c1) * NT + m] = s1 + 64.f * p.bias[t][c1];
  }
}

__global__ __launch_bounds__(256) void k2b_reduce(const float* partial, float* gcol) {
  __shared__ float red[256];
  const int b = blockIdx.x, tid = threadIdx.x;   // b = t*256 + col
  const float* p = partial + (size_t)b * NT;
  float s = 0.f;
#pragma unroll
  for (int i = 0; i < 4; ++i) s += p[i * 256 + tid];
  red[tid] = s;
  __syncthreads();
  for (int st = 128; st > 0; st >>= 1) {
    if (tid < st) red[tid] += red[tid + st];
    __syncthreads();
  }
  if (tid == 0) gcol[b] = red[0];
}

// ---------------------------------------------------------------------------
struct P3 { const float* gcol; const float* Wqkv; const float* bqkv; float* km; float* vm; };

__global__ __launch_bounds__(256) void k3_means(P3 p) {
  __shared__ float mn[5][256];
  const int d = threadIdx.x;
  for (int t = 0; t < 5; ++t) mn[t][d] = p.gcol[t * 256 + d] * (1.0f / 65536.0f);
  __syncthreads();
  const float* Wk = p.Wqkv + 256 * 256;
  const float* Wv = p.Wqkv + 512 * 256;
  const float bk = p.bqkv[256 + d], bv = p.bqkv[512 + d];
  for (int t = 0; t < 5; ++t) {
    float sk = 0.f, sv = 0.f;
    for (int k = 0; k < 256; ++k) {
      const float mv = mn[t][k];
      sk += mv * Wk[d * 256 + k];
      sv += mv * Wv[d * 256 + k];
    }
    p.km[t * 256 + d] = sk + bk;
    p.vm[t * 256 + d] = sv + bv;
  }
}

// ---------------------------------------------------------------------------
struct P4 {
  const float* x[5]; const float* bias[5];
  const __bf16* wt;                 // [9][65536]: 0..4 per-type, 5 Wq, 6 Wk, 7 Wv, 8 Wo
  const float* bqkv; const float* bo;
  const float* km; const float* vm; // [5][256]
  const float* gamma; const float* beta;
  float* out;
};

__global__ __launch_bounds__(256) void k4_attn(P4 p) {
  // 224 KB LDS (ISA: up to 320 KB per WG), regions reused across stages:
  //   0..64K   projf (f32)     | live: proj -> residual
  //  64..96K   projb (bf16)    | live: proj -> q/k/v GEMMs
  //  96..128K  xb (bf16)       | stage -> proj GEMM, reused as att afterwards
  // 128..160K  qs (bf16)       | q GEMM -> attention
  // 160..192K  ks (bf16)       | k GEMM -> attention
  // 192..224K  vs (bf16)       | v GEMM -> attention
  // 128..192K  xf32 stage (pre-proj) and outf f32 (post-attention) overlays
  __shared__ __attribute__((aligned(16))) unsigned char smem[224 * 1024];
  float*  projf = (float*)smem;
  __bf16* projb = (__bf16*)(smem + 64 * 1024);
  __bf16* xb    = (__bf16*)(smem + 96 * 1024);
  __bf16* att   = xb;
  __bf16* qs    = (__bf16*)(smem + 128 * 1024);
  __bf16* ks    = (__bf16*)(smem + 160 * 1024);
  __bf16* vs    = (__bf16*)(smem + 192 * 1024);
  float*  xf    = (float*)(smem + 128 * 1024);
  float*  outf  = (float*)(smem + 128 * 1024);

  const int tid = threadIdx.x, m = blockIdx.x, t = blockIdx.y + 2;
  const int lane = tid & 31, wave = tid >> 5;
  const int half = lane >> 4, l16 = lane & 15, ct0 = wave * 2;
  const int col0 = ct0 * 16 + l16, col1 = col0 + 16;

  stage_x64(p.x[t] + (size_t)m * 64 * 256, xb, xf, tid);
  __syncthreads();

  // ---- proj = x @ W_t^T + b_t (keep f32 + bf16 copies in LDS) ----
  {
    v8f acc[4][2] = {};
    gemm_tile64(xb, p.wt + (size_t)t * 65536, l16, half, ct0, acc);
    const float b0 = p.bias[t][col0], b1 = p.bias[t][col1];
#pragma unroll
    for (int rt = 0; rt < 4; ++rt)
#pragma unroll
      for (int j = 0; j < 8; ++j) {
        const float v0 = acc[rt][0][j] + b0, v1 = acc[rt][1][j] + b1;
        const int r = (rt * 16 + j + half * 8) * 256;
        projf[r + col0] = v0;          projf[r + col1] = v1;
        projb[r + col0] = (__bf16)v0;  projb[r + col1] = (__bf16)v1;
      }
  }
  __syncthreads();

  // ---- q, k(self), v(self) GEMMs -> LDS (bf16) ----
  {
    v8f acc[4][2] = {};
    gemm_tile64(projb, p.wt + 5ull * 65536, l16, half, ct0, acc);
    const float b0 = p.bqkv[col0], b1 = p.bqkv[col1];
#pragma unroll
    for (int rt = 0; rt < 4; ++rt)
#pragma unroll
      for (int j = 0; j < 8; ++j) {
        const int r = (rt * 16 + j + half * 8) * 256;
        qs[r + col0] = (__bf16)(acc[rt][0][j] + b0);
        qs[r + col1] = (__bf16)(acc[rt][1][j] + b1);
      }
  }
  {
    v8f acc[4][2] = {};
    gemm_tile64(projb, p.wt + 6ull * 65536, l16, half, ct0, acc);
    const float b0 = p.bqkv[256 + col0], b1 = p.bqkv[256 + col1];
#pragma unroll
    for (int rt = 0; rt < 4; ++rt)
#pragma unroll
      for (int j = 0; j < 8; ++j) {
        const int r = (rt * 16 + j + half * 8) * 256;
        ks[r + col0] = (__bf16)(acc[rt][0][j] + b0);
        ks[r + col1] = (__bf16)(acc[rt][1][j] + b1);
      }
  }
  {
    v8f acc[4][2] = {};
    gemm_tile64(projb, p.wt + 7ull * 65536, l16, half, ct0, acc);
    const float b0 = p.bqkv[512 + col0], b1 = p.bqkv[512 + col1];
#pragma unroll
    for (int rt = 0; rt < 4; ++rt)
#pragma unroll
      for (int j = 0; j < 8; ++j) {
        const int r = (rt * 16 + j + half * 8) * 256;
        vs[r + col0] = (__bf16)(acc[rt][0][j] + b0);
        vs[r + col1] = (__bf16)(acc[rt][1][j] + b1);
      }
  }
  __syncthreads();

  // ---- per-row attention: one (row, head) per thread, L<=4 neighbors ----
  {
    const int row = tid >> 2, h = tid & 3;
    const int dbase = h * 64;
    const __bf16* qrow = qs + row * 256 + dbase;

    int L, src[3];
    if (t == 2)      { L = 3; src[0] = 0; src[1] = 1; src[2] = 0; }  // query <- user,llm
    else if (t == 3) { L = 2; src[0] = 2; src[1] = 0; src[2] = 0; }  // response <- query
    else             { L = 4; src[0] = 3; src[1] = 2; src[2] = 0; }  // turn <- resp,query,user

    float sc[4];
    {
      const __bf16* krow = ks + row * 256 + dbase;
      float s = 0.f;
      for (int i = 0; i < 64; ++i) s += (float)qrow[i] * (float)krow[i];
      sc[0] = s;
    }
#pragma unroll
    for (int l = 1; l < 4; ++l) {
      float s = 0.f;
      if (l < L) {
        const float* kmv = p.km + src[l - 1] * 256 + dbase;
        for (int i = 0; i < 64; ++i) s += (float)qrow[i] * kmv[i];
      }
      sc[l] = s;
    }
    const float scale = 0.125f;              // 1/sqrt(64)
    float mx = -1e30f;
#pragma unroll
    for (int l = 0; l < 4; ++l) { sc[l] *= scale; if (l < L && sc[l] > mx) mx = sc[l]; }
    float ex[4], den = 0.f;
#pragma unroll
    for (int l = 0; l < 4; ++l) { ex[l] = (l < L) ? __expf(sc[l] - mx) : 0.f; den += ex[l]; }
    const float inv = 1.f / den;
    const float w0 = ex[0] * inv;
    const float w1 = ex[1] * inv, w2 = ex[2] * inv, w3 = ex[3] * inv;
    const float* vm1 = p.vm + src[0] * 256 + dbase;
    const float* vm2 = p.vm + src[1] * 256 + dbase;
    const float* vm3 = p.vm + src[2] * 256 + dbase;
    const __bf16* vrow = vs + row * 256 + dbase;
    __bf16* arow = att + row * 256 + dbase;
    for (int i = 0; i < 64; ++i) {
      float o = w0 * (float)vrow[i] + w1 * vm1[i] + w2 * vm2[i] + w3 * vm3[i];
      arow[i] = (__bf16)o;
    }
  }
  __syncthreads();

  // ---- out = att @ Wo^T + bo + proj (residual) ----
  {
    v8f acc[4][2] = {};
    gemm_tile64(att, p.wt + 8ull * 65536, l16, half, ct0, acc);
    const float b0 = p.bo[col0], b1 = p.bo[col1];
#pragma unroll
    for (int rt = 0; rt < 4; ++rt)
#pragma unroll
      for (int j = 0; j < 8; ++j) {
        const int r = (rt * 16 + j + half * 8) * 256;
        outf[r + col0] = acc[rt][0][j] + b0 + projf[r + col0];
        outf[r + col1] = acc[rt][1][j] + b1 + projf[r + col1];
      }
  }
  __syncthreads();

  // ---- LayerNorm per row (4 lanes per row) + write out ----
  {
    const int row = tid >> 2, seg = tid & 3;
    const float* orow = outf + row * 256 + seg * 64;
    float s = 0.f, q2 = 0.f;
    for (int i = 0; i < 64; ++i) { const float v = orow[i]; s += v; q2 += v * v; }
    s += __shfl_xor(s, 1);  q2 += __shfl_xor(q2, 1);
    s += __shfl_xor(s, 2);  q2 += __shfl_xor(q2, 2);
    const float mu  = s * (1.f / 256.f);
    const float var = q2 * (1.f / 256.f) - mu * mu;
    const float rin = rsqrtf(var + 1e-5f);
    float* outp = p.out + ((size_t)t * NROW + (size_t)m * 64 + row) * 256 + seg * 64;
    const float* g = p.gamma + seg * 64;
    const float* be = p.beta + seg * 64;
    for (int i = 0; i < 64; ++i)
      outp[i] = (orow[i] - mu) * rin * g[i] + be[i];
  }
}

// ---------------------------------------------------------------------------
// user/llm: out = LN(2 * proj)
__global__ __launch_bounds__(256) void k5_simple(P4 p) {
  __shared__ __attribute__((aligned(16))) unsigned char smem[96 * 1024];
  __bf16* xb   = (__bf16*)smem;               // 32 KB
  float*  xf   = (float*)(smem + 32 * 1024);  // 64 KB stage, overlaid by outf
  float*  outf = (float*)(smem + 32 * 1024);  // 64 KB

  const int tid = threadIdx.x, m = blockIdx.x, t = blockIdx.y;  // t = 0,1
  const int lane = tid & 31, wave = tid >> 5;
  const int half = lane >> 4, l16 = lane & 15, ct0 = wave * 2;
  const int col0 = ct0 * 16 + l16, col1 = col0 + 16;

  stage_x64(p.x[t] + (size_t)m * 64 * 256, xb, xf, tid);
  __syncthreads();

  v8f acc[4][2] = {};
  gemm_tile64(xb, p.wt + (size_t)t * 65536, l16, half, ct0, acc);
  const float b0 = p.bias[t][col0], b1 = p.bias[t][col1];
#pragma unroll
  for (int rt = 0; rt < 4; ++rt)
#pragma unroll
    for (int j = 0; j < 8; ++j) {
      const int r = (rt * 16 + j + half * 8) * 256;
      outf[r + col0] = 2.f * (acc[rt][0][j] + b0);
      outf[r + col1] = 2.f * (acc[rt][1][j] + b1);
    }
  __syncthreads();

  const int row = tid >> 2, seg = tid & 3;
  const float* orow = outf + row * 256 + seg * 64;
  float s = 0.f, q2 = 0.f;
  for (int i = 0; i < 64; ++i) { const float v = orow[i]; s += v; q2 += v * v; }
  s += __shfl_xor(s, 1);  q2 += __shfl_xor(q2, 1);
  s += __shfl_xor(s, 2);  q2 += __shfl_xor(q2, 2);
  const float mu  = s * (1.f / 256.f);
  const float var = q2 * (1.f / 256.f) - mu * mu;
  const float rin = rsqrtf(var + 1e-5f);
  float* outp = p.out + ((size_t)t * NROW + (size_t)m * 64 + row) * 256 + seg * 64;
  const float* g = p.gamma + seg * 64;
  const float* be = p.beta + seg * 64;
  for (int i = 0; i < 64; ++i)
    outp[i] = (orow[i] - mu) * rin * g[i] + be[i];
}

// ---------------------------------------------------------------------------
extern "C" void kernel_launch(void* const* d_in, const int* in_sizes, int n_in,
                              void* d_out, int out_size, void* d_ws, size_t ws_size,
                              hipStream_t stream) {
  const float* xs[5] = {(const float*)d_in[0], (const float*)d_in[3],
                        (const float*)d_in[6], (const float*)d_in[9],
                        (const float*)d_in[12]};
  const float* Wsrc[5] = {(const float*)d_in[1], (const float*)d_in[4],
                          (const float*)d_in[7], (const float*)d_in[10],
                          (const float*)d_in[13]};
  const float* bs[5] = {(const float*)d_in[2], (const float*)d_in[5],
                        (const float*)d_in[8], (const float*)d_in[11],
                        (const float*)d_in[14]};
  const float* Wqkv  = (const float*)d_in[15];
  const float* bqkv  = (const float*)d_in[16];
  const float* Wo    = (const float*)d_in[17];
  const float* bo    = (const float*)d_in[18];
  const float* gamma = (const float*)d_in[19];
  const float* beta  = (const float*)d_in[20];

  // workspace layout
  const size_t OFF_WBF  = 0;                           // 9*65536*2 = 1179648 B
  const size_t OFF_GCOL = 1179648;                     // 1280 f32
  const size_t OFF_KM   = OFF_GCOL + 5120;             // 1280 f32
  const size_t OFF_VM   = OFF_KM + 5120;               // 1280 f32
  const size_t OFF_PART = OFF_VM + 5120;               // 5*256*1024 f32
  const size_t WS_NEED  = OFF_PART + (size_t)5 * 256 * NT * 4;
  if (ws_size < WS_NEED) return;

  char* ws = (char*)d_ws;
  __bf16* wbf    = (__bf16*)(ws + OFF_WBF);
  float* gcol    = (float*)(ws + OFF_GCOL);
  float* km      = (float*)(ws + OFF_KM);
  float* vm      = (float*)(ws + OFF_VM);
  float* partial = (float*)(ws + OFF_PART);

  // k1: convert 9 weight matrices to bf16
  P1 p1;
  p1.src[0] = Wsrc[0]; p1.src[1] = Wsrc[1]; p1.src[2] = Wsrc[2];
  p1.src[3] = Wsrc[3]; p1.src[4] = Wsrc[4];
  p1.src[5] = Wqkv;                // Wq
  p1.src[6] = Wqkv + 256 * 256;    // Wk
  p1.src[7] = Wqkv + 512 * 256;    // Wv
  p1.src[8] = Wo;
  p1.dst = wbf;
  k1_cvtw<<<2304, 256, 0, stream>>>(p1);

  // k2 + k2b: deterministic per-type column sums of proj
  P2 p2;
  for (int i = 0; i < 5; ++i) { p2.x[i] = xs[i]; p2.bias[i] = bs[i]; }
  p2.wt = wbf; p2.partial = partial;
  k2_projsum<<<dim3(NT, 5), 256, 0, stream>>>(p2);
  k2b_reduce<<<1280, 256, 0, stream>>>(partial, gcol);

  // k3: means -> km/vm vectors
  P3 p3; p3.gcol = gcol; p3.Wqkv = Wqkv; p3.bqkv = bqkv; p3.km = km; p3.vm = vm;
  k3_means<<<1, 256, 0, stream>>>(p3);

  // k4/k5: fused per-type pipelines
  P4 p4;
  for (int i = 0; i < 5; ++i) { p4.x[i] = xs[i]; p4.bias[i] = bs[i]; }
  p4.wt = wbf; p4.bqkv = bqkv; p4.bo = bo; p4.km = km; p4.vm = vm;
  p4.gamma = gamma; p4.beta = beta; p4.out = (float*)d_out;
  k4_attn<<<dim3(NT, 3), 256, 0, stream>>>(p4);
  k5_simple<<<dim3(NT, 2), 256, 0, stream>>>(p4);
}
